// Attention_15925738733936
// MI455X (gfx1250) — matrix operand order repaired and checked by reading, not compiled, not run on hardware
//
#include <hip/hip_runtime.h>
#include <hip/hip_bf16.h>
#include <math.h>

// ---------------------------------------------------------------------------
// Problem constants: B=1, S=2048, D=4096, H=32, KVH=8, HD=128
// ---------------------------------------------------------------------------
#define S_LEN 2048
#define DMODEL 4096
#define NHEAD 32
#define NKVHEAD 8
#define HEADDIM 128
#define NQ (NHEAD * HEADDIM)      // 4096
#define NKV (NKVHEAD * HEADDIM)   // 1024
#define NEG_BIG (-1.0e9f)

typedef __bf16 bf16_t;
typedef __attribute__((ext_vector_type(16))) __bf16 v16bf;
typedef __attribute__((ext_vector_type(8)))  __bf16 v8bf;
typedef __attribute__((ext_vector_type(8)))  float  v8f;
typedef __attribute__((ext_vector_type(4)))  unsigned int u32x4;
typedef __attribute__((ext_vector_type(8)))  int i32x8;
typedef __attribute__((ext_vector_type(4)))  int i32x4;

#if __has_builtin(__builtin_amdgcn_tensor_load_to_lds) && \
    __has_builtin(__builtin_amdgcn_s_wait_tensorcnt)
#define USE_TDM 1
#endif

static __device__ __forceinline__ v8f wmma_bf16(v16bf a, v16bf b, v8f c) {
  return __builtin_amdgcn_wmma_f32_16x16x32_bf16(
      false, a, false, b, (short)0, c, false, false);
}

// Assemble a 16-element bf16 A/B fragment from two aligned 16B LDS chunks.
static __device__ __forceinline__ v16bf frag_from(const bf16_t* p0, const bf16_t* p1) {
  v8bf a = *(const v8bf*)p0;
  v8bf b = *(const v8bf*)p1;
  v16bf r;
#pragma unroll
  for (int j = 0; j < 8; ++j) { r[j] = a[j]; r[8 + j] = b[j]; }
  return r;
}

// 8-element contiguous load + convert-to-bf16 into LDS (overloads avoid if-constexpr)
static __device__ __forceinline__ void load8_cvt(const float* p, bf16_t* d) {
  const float4* q = (const float4*)p;
  float4 a = q[0], b = q[1];
  d[0] = (__bf16)a.x; d[1] = (__bf16)a.y; d[2] = (__bf16)a.z; d[3] = (__bf16)a.w;
  d[4] = (__bf16)b.x; d[5] = (__bf16)b.y; d[6] = (__bf16)b.z; d[7] = (__bf16)b.w;
}
static __device__ __forceinline__ void load8_cvt(const bf16_t* p, bf16_t* d) {
  v8bf u = *(const v8bf*)p;
#pragma unroll
  for (int j = 0; j < 8; ++j) d[j] = u[j];
}

// ---------------------------------------------------------------------------
// Tiled GEMM: C[M,N] = A[M,K] @ B_f32[K,N] with bf16 WMMA, fp32 accumulate.
// Block 256 threads = 8 waves; block tile 64x64, K-step 32.
// LDS: As[64][40] row-major (80B rows, 16B aligned); Bt[64][40] = B transposed
// per column so every fragment read is two aligned ds_load_b128.
// Wave w: rows (w&3)*16.., cols (w>>2)*32.. (2 accumulators, 2 WMMA / K-step).
// ---------------------------------------------------------------------------
template <typename AT, typename CT>
__global__ __launch_bounds__(256)
void gemm_tiled(const AT* __restrict__ A, const float* __restrict__ B,
                CT* __restrict__ C, int M, int N, int K) {
  __shared__ bf16_t As[64][40];
  __shared__ bf16_t Bt[64][40];   // [col][k]

  const int tid  = threadIdx.x;
  const int lane = tid & 31;
  const int wave = tid >> 5;
  const int l16  = lane & 15;
  const int hi   = lane >> 4;
  const int mSub = wave & 3;
  const int nSub = wave >> 2;
  const size_t tileM = (size_t)blockIdx.y * 64;
  const size_t tileN = (size_t)blockIdx.x * 64;

  // staging roles
  const int ar = tid >> 2, aseg = tid & 3;   // A: 64 rows x (4 x 8 cols)
  const int bk = tid >> 3, bseg = tid & 7;   // B: 32 k-rows x (8 x 8 cols)
  const AT*    Ag = A + (tileM + ar) * (size_t)K + aseg * 8;
  const float* Bg = B + (size_t)bk * N + tileN + bseg * 8;

  v8f acc[2] = {(v8f){}, (v8f){}};

  for (int kk = 0; kk < K; kk += 32) {
    // ---- cooperative staging (contiguous global loads, bf16 into LDS) ----
    load8_cvt(Ag, &As[ar][aseg * 8]);
    {
      const float4* q = (const float4*)Bg;
      float4 b0 = q[0], b1 = q[1];
      Bt[bseg * 8 + 0][bk] = (__bf16)b0.x;
      Bt[bseg * 8 + 1][bk] = (__bf16)b0.y;
      Bt[bseg * 8 + 2][bk] = (__bf16)b0.z;
      Bt[bseg * 8 + 3][bk] = (__bf16)b0.w;
      Bt[bseg * 8 + 4][bk] = (__bf16)b1.x;
      Bt[bseg * 8 + 5][bk] = (__bf16)b1.y;
      Bt[bseg * 8 + 6][bk] = (__bf16)b1.z;
      Bt[bseg * 8 + 7][bk] = (__bf16)b1.w;
    }
    // prefetch next K tile while this one is consumed
    __builtin_prefetch(Ag + 32, 0, 3);
    __builtin_prefetch(Bg + (size_t)32 * N, 0, 3);
    __syncthreads();

    // ---- wave-level WMMA on the LDS tiles ----
    const bf16_t* arow = &As[mSub * 16 + l16][0];
    v16bf a = frag_from(arow + hi * 8, arow + hi * 8 + 16);
#pragma unroll
    for (int t = 0; t < 2; ++t) {
      const bf16_t* bcol = &Bt[nSub * 32 + t * 16 + l16][0];
      v16bf b = frag_from(bcol + hi * 8, bcol + hi * 8 + 16);
      acc[t] = wmma_bf16(a, b, acc[t]);
    }
    __syncthreads();
    Ag += 32;
    Bg += (size_t)32 * N;
  }

#pragma unroll
  for (int t = 0; t < 2; ++t) {
    const size_t col = tileN + nSub * 32 + t * 16 + l16;
#pragma unroll
    for (int r = 0; r < 8; ++r) {
      const size_t row = tileM + mSub * 16 + r + hi * 8;
      C[row * N + col] = (CT)acc[t][r];
    }
  }
}

// ---------------------------------------------------------------------------
// RoPE: in-place on bf16 Q [S,H,HD] and K [S,KVH,HD]. One thread per pair.
// ---------------------------------------------------------------------------
__global__ void rope_kernel(bf16_t* __restrict__ q, bf16_t* __restrict__ k,
                            const float* __restrict__ fcos,
                            const float* __restrict__ fsin) {
  const long long i = (long long)blockIdx.x * blockDim.x + threadIdx.x;
  const long long qpairs = (long long)S_LEN * NHEAD * (HEADDIM / 2);
  const long long kpairs = (long long)S_LEN * NKVHEAD * (HEADDIM / 2);
  if (i >= qpairs + kpairs) return;

  bf16_t* t; int nh; long long j;
  if (i < qpairs) { t = q; nh = NHEAD;   j = i; }
  else            { t = k; nh = NKVHEAD; j = i - qpairs; }

  const int p = (int)(j % (HEADDIM / 2));
  long long rest = j / (HEADDIM / 2);
  const int h = (int)(rest % nh);
  const long long s = rest / nh;

  const float c  = fcos[s * (HEADDIM / 2) + p];
  const float sn = fsin[s * (HEADDIM / 2) + p];

  bf16_t* base = t + ((size_t)s * nh + h) * HEADDIM + 2 * p;
  const float t0 = (float)base[0];
  const float t1 = (float)base[1];
  base[0] = (__bf16)(t0 * c - t1 * sn);
  base[1] = (__bf16)(t0 * sn + t1 * c);
}

#ifdef USE_TDM
// TDM: async copy one 32x128 bf16 K tile (row stride NKV elems) into LDS with
// row padding 128->136 elems (pad_interval=64 DWORDs, pad_amount=4 DWORDs).
// D# per CDNA5 ISA 8.3/8.4. This toolchain exposes the 6-arg builtin:
//   (uint32x4 g0, int32x8 g1, int32x4 g2, int32x4 g3, int32x8 extra, i32 cpol)
// Groups 2/3 (and the extra group) are zero for a 2D tile.
static __device__ __forceinline__ void tdm_load_k_tile(const bf16_t* gsrc,
                                                       void* lds_dst) {
  const unsigned long long ga = (unsigned long long)(size_t)gsrc;
  u32x4 g0;
  g0[0] = 1u;                                        // count=1 valid descriptor
  g0[1] = (unsigned int)(size_t)lds_dst;             // lds_addr (byte offset)
  g0[2] = (unsigned int)(ga & 0xFFFFFFFFu);          // global_addr[31:0]
  g0[3] = (unsigned int)((ga >> 32) & 0x01FFFFFFu) | (2u << 30); // [56:32]|type=2
  i32x8 g1;
  g1[0] = (1 << 16)        // data_size = 1 -> 2 bytes
        | (1 << 20)        // pad_enable
        | (5 << 22)        // pad_interval code 5 = 64 DWORDs (one 128-elem row)
        | (3 << 25);       // pad_amount  code 3 = 4 DWORDs (8 bf16)
  g1[1] = (HEADDIM & 0xFFFF) << 16;                  // tensor_dim0[15:0]
  g1[2] = (S_LEN & 0xFFFF) << 16;                    // tensor_dim1[15:0]
  g1[3] = (HEADDIM << 16);                           // tile_dim0 = 128
  g1[4] = 32;                                        // tile_dim1 = 32
  g1[5] = NKV;                                       // tensor_dim0_stride
  g1[6] = 0;
  g1[7] = 0;
  const i32x4 z4 = {0, 0, 0, 0};
  const i32x8 z8 = {0, 0, 0, 0, 0, 0, 0, 0};
  __builtin_amdgcn_tensor_load_to_lds(g0, g1, z4, z4, z8, 0);
}
#endif

// ---------------------------------------------------------------------------
// Flash attention (causal, GQA 4:1). Block = 128 threads = 4 waves; block owns
// one 64-row Q tile of one head (wave w -> rows w*16..). Key loop is block-
// uniform so K/V tiles are cooperatively staged in LDS:
//   Ks[32][136]  row-major K tile (TDM async copy when available)
//   Vt[128][40]  transposed V tile -> aligned b128 fragment reads
// Online softmax with shfl reductions; QK^T and PV on v_wmma bf16.
// ---------------------------------------------------------------------------
__global__ __launch_bounds__(128, 1)
void flash_attn(const bf16_t* __restrict__ q, const bf16_t* __restrict__ k,
                const bf16_t* __restrict__ v, bf16_t* __restrict__ attn) {
  __shared__ bf16_t Ks[32][136];
  __shared__ bf16_t Vt[128][40];
  __shared__ float  Pl[4][16 * 32];

  const int tid  = threadIdx.x;
  const int lane = tid & 31;
  const int wave = tid >> 5;
  const int l16  = lane & 15;
  const int hi   = lane >> 4;

  const int h     = blockIdx.y;
  const int kvh   = h >> 2;                 // H/KVH = 4
  const int qTile = blockIdx.x * 64;
  const int qBase = qTile + wave * 16;
  const float scale = 0.08838834764831845f; // 1/sqrt(128)

  // ---- per-wave Q fragments (4 K-chunks of 32 along HD) ----
  v16bf qa[4];
  {
    const bf16_t* qrow = q + (size_t)(qBase + l16) * NQ + (size_t)h * HEADDIM;
#pragma unroll
    for (int i = 0; i < 4; ++i) {
      const int base = i * 32 + hi * 8;
#pragma unroll
      for (int j = 0; j < 8; ++j) qa[i][j] = qrow[base + j];
#pragma unroll
      for (int j = 0; j < 8; ++j) qa[i][8 + j] = qrow[base + 16 + j];
    }
  }

  float m[8], l[8];
  v8f acc[8];
#pragma unroll
  for (int r = 0; r < 8; ++r) { m[r] = -1.0e30f; l[r] = 0.0f; }
#pragma unroll
  for (int t = 0; t < 8; ++t) acc[t] = (v8f){};

  float* myp = &Pl[wave][0];
  const int sr = tid >> 2, sseg = tid & 3;  // staging: key-row, 32-col segment

  for (int keyBase = 0; keyBase <= qTile + 63; keyBase += 32) {
    // ---- stage V transposed (all threads) ----
    {
      int key = keyBase + sr; if (key >= S_LEN) key = S_LEN - 1;
      const bf16_t* vrow = v + (size_t)key * NKV + (size_t)kvh * HEADDIM + sseg * 32;
#pragma unroll
      for (int j = 0; j < 32; ++j) Vt[sseg * 32 + j][sr] = vrow[j];
    }
    // ---- stage K tile ----
#ifdef USE_TDM
    if (wave == 0) {
      tdm_load_k_tile(k + (size_t)keyBase * NKV + (size_t)kvh * HEADDIM, &Ks[0][0]);
      __builtin_amdgcn_s_wait_tensorcnt(0);
    }
#else
    {
      int key = keyBase + sr; if (key >= S_LEN) key = S_LEN - 1;
      const bf16_t* krow = k + (size_t)key * NKV + (size_t)kvh * HEADDIM + sseg * 32;
#pragma unroll
      for (int j = 0; j < 4; ++j)
        *(v8bf*)&Ks[sr][sseg * 32 + j * 8] = *(const v8bf*)(krow + j * 8);
    }
#endif
    __syncthreads();

    // ---- scores for the two 16-key halves ----
    float p0[8], p1[8];
#pragma unroll
    for (int u = 0; u < 2; ++u) {
      const int keyIdx = keyBase + u * 16 + l16;
      const bf16_t* krow = &Ks[u * 16 + l16][0];
      v8f s = {};
#pragma unroll
      for (int i = 0; i < 4; ++i) {
        const int base = i * 32 + hi * 8;
        v16bf kb = frag_from(krow + base, krow + base + 16);
        s = wmma_bf16(qa[i], kb, s);
      }
      float* dst = u ? p1 : p0;
#pragma unroll
      for (int r = 0; r < 8; ++r) {
        const int row = qBase + r + hi * 8;
        float val = s[r] * scale;
        if (keyIdx > row) val = NEG_BIG;   // causal (also kills OOB/clamped)
        dst[r] = val;
      }
    }

    // ---- online softmax update (reductions across 16-lane groups) ----
    float rmax[8];
#pragma unroll
    for (int r = 0; r < 8; ++r) rmax[r] = fmaxf(p0[r], p1[r]);
#pragma unroll
    for (int off = 1; off < 16; off <<= 1)
#pragma unroll
      for (int r = 0; r < 8; ++r)
        rmax[r] = fmaxf(rmax[r], __shfl_xor(rmax[r], off, 32));

    float rsum[8];
#pragma unroll
    for (int r = 0; r < 8; ++r) {
      const float mnew = fmaxf(m[r], rmax[r]);
      const float corr = __expf(m[r] - mnew);
      p0[r] = __expf(p0[r] - mnew);
      p1[r] = __expf(p1[r] - mnew);
      rsum[r] = p0[r] + p1[r];
      l[r] *= corr;
      m[r] = mnew;
#pragma unroll
      for (int t = 0; t < 8; ++t) acc[t][r] *= corr;
    }
#pragma unroll
    for (int off = 1; off < 16; off <<= 1)
#pragma unroll
      for (int r = 0; r < 8; ++r)
        rsum[r] += __shfl_xor(rsum[r], off, 32);
#pragma unroll
    for (int r = 0; r < 8; ++r) l[r] += rsum[r];

    // ---- transpose P (C-layout -> A-layout) via per-wave LDS ----
#pragma unroll
    for (int r = 0; r < 8; ++r) {
      myp[(r + hi * 8) * 32 + l16]      = p0[r];
      myp[(r + hi * 8) * 32 + 16 + l16] = p1[r];
    }
    asm volatile("s_wait_dscnt 0" ::: "memory"); // per-wave LDS RAW fence

    v16bf pa;
#pragma unroll
    for (int j = 0; j < 8; ++j) pa[j]     = (__bf16)myp[l16 * 32 + hi * 8 + j];
#pragma unroll
    for (int j = 0; j < 8; ++j) pa[8 + j] = (__bf16)myp[l16 * 32 + 16 + hi * 8 + j];

    // ---- PV: 8 WMMAs over HD chunks of 16, from transposed V in LDS ----
#pragma unroll
    for (int t = 0; t < 8; ++t) {
      const bf16_t* vcol = &Vt[t * 16 + l16][0];
      v16bf vb = frag_from(vcol + hi * 8, vcol + hi * 8 + 16);
      acc[t] = wmma_bf16(pa, vb, acc[t]);
    }
    __syncthreads();   // protect LDS tiles before restaging
  }

  // ---- normalize and store attention output [S, H*HD] as bf16 ----
#pragma unroll
  for (int t = 0; t < 8; ++t) {
#pragma unroll
    for (int r = 0; r < 8; ++r) {
      const int row = qBase + r + hi * 8;
      const int col = h * HEADDIM + t * 16 + l16;
      attn[(size_t)row * NQ + col] = (__bf16)(acc[t][r] / l[r]);
    }
  }
}

// ---------------------------------------------------------------------------
// Host-side launch
// ---------------------------------------------------------------------------
extern "C" void kernel_launch(void* const* d_in, const int* in_sizes, int n_in,
                              void* d_out, int out_size, void* d_ws, size_t ws_size,
                              hipStream_t stream) {
  const float* x    = (const float*)d_in[0];
  const float* wq   = (const float*)d_in[1];
  const float* wk   = (const float*)d_in[2];
  const float* wv   = (const float*)d_in[3];
  const float* wo   = (const float*)d_in[4];
  const float* fcos = (const float*)d_in[5];
  const float* fsin = (const float*)d_in[6];
  float* out = (float*)d_out;

  // Workspace (bf16): q[S,NQ] | k[S,NKV] | v[S,NKV] | attn[S,NQ]  (~42 MB)
  bf16_t* qb   = (bf16_t*)d_ws;
  bf16_t* kb   = qb + (size_t)S_LEN * NQ;
  bf16_t* vb   = kb + (size_t)S_LEN * NKV;
  bf16_t* attn = vb + (size_t)S_LEN * NKV;

  // 1) QKV projections (fp32 in -> bf16 out, fp32 WMMA accumulate)
  gemm_tiled<float, bf16_t><<<dim3(NQ / 64,  S_LEN / 64), 256, 0, stream>>>(x, wq, qb, S_LEN, NQ,  DMODEL);
  gemm_tiled<float, bf16_t><<<dim3(NKV / 64, S_LEN / 64), 256, 0, stream>>>(x, wk, kb, S_LEN, NKV, DMODEL);
  gemm_tiled<float, bf16_t><<<dim3(NKV / 64, S_LEN / 64), 256, 0, stream>>>(x, wv, vb, S_LEN, NKV, DMODEL);

  // 2) RoPE on q and k (in place)
  {
    const long long pairs = (long long)S_LEN * (NHEAD + NKVHEAD) * (HEADDIM / 2);
    rope_kernel<<<(int)((pairs + 255) / 256), 256, 0, stream>>>(qb, kb, fcos, fsin);
  }

  // 3) causal GQA flash attention
  flash_attn<<<dim3(S_LEN / 64, NHEAD), 128, 0, stream>>>(qb, kb, vb, attn);

  // 4) output projection (bf16 x fp32 -> fp32)
  gemm_tiled<bf16_t, float><<<dim3(DMODEL / 64, S_LEN / 64), 256, 0, stream>>>(attn, wo, out, S_LEN, DMODEL, NQ);
}
